// CorrelationMapLayer_28956669510096
// MI455X (gfx1250) — matrix operand order repaired and checked by reading, not compiled, not run on hardware
//
#include <hip/hip_runtime.h>
#include <hip/hip_bf16.h>

// ---------------------------------------------------------------------------
// Types for CDNA5 WMMA (gfx1250, wave32)
// ---------------------------------------------------------------------------
typedef __attribute__((ext_vector_type(16))) __bf16 v16bf;
typedef __attribute__((ext_vector_type(8)))  __bf16 v8bf;
typedef __attribute__((ext_vector_type(8)))  float  v8f;

#define B_   32
#define C_   256
#define HS   56
#define OH   28
#define PQ   784      // 28*28
#define P2   800      // padded to multiple of 32 (K and M/N tiling)
#define KT1  25       // 800/32 K-tiles for the 1x1 convs
#define KTC  8        // 256/32 K-tiles for the correlation
#define OC_ALL 512

// ---------------------------------------------------------------------------
// WMMA fragment helpers (A/B use contiguous k order k=16h+s, same in both; wave32)
// ---------------------------------------------------------------------------
// A (row-major M x K): lane<16 -> row=lane, K=k0..k0+15
//                      lane>=16 -> row=lane-16, K=k0+16..k0+31
__device__ __forceinline__ v16bf load_a_frag(const __bf16* A, int lda,
                                             int row_base, int k0, int lane) {
    int r  = row_base + (lane & 15);
    int kh = k0 + ((lane >> 4) << 4);          // +0 or +16
    const __bf16* p0 = A + (size_t)r * lda + kh;
    v8bf lo = *(const v8bf*)p0;                // 16B aligned
    v8bf hi = *(const v8bf*)(p0 + 8);
    v16bf a;
#pragma unroll
    for (int i = 0; i < 8; ++i) { a[i] = lo[i]; a[i + 8] = hi[i]; }
    return a;
}

// B operand pre-swizzled: S[(k/32)][n][k%32]  (column's 32 K-values contiguous)
// fragment: lane<16 -> col=n+lane, K=k0..k0+15 ; lane>=16 -> col, K=k0+16..k0+31
// => ONE contiguous 32-byte load per lane.
__device__ __forceinline__ v16bf load_b_sw(const __bf16* S, int ncols,
                                           int k0, int col_base, int lane) {
    const __bf16* p = S + ((size_t)(k0 >> 5) * ncols + (col_base + (lane & 15))) * 32
                        + ((lane >> 4) << 4);
    v8bf lo = *(const v8bf*)p;
    v8bf hi = *(const v8bf*)(p + 8);
    v16bf b;
#pragma unroll
    for (int i = 0; i < 8; ++i) { b[i] = lo[i]; b[i + 8] = hi[i]; }
    return b;
}

__device__ __forceinline__ v8f wmma_bf16(v16bf a, v16bf b, v8f c) {
    return __builtin_amdgcn_wmma_f32_16x16x32_bf16(
        false, a, false, b, (short)0, c, false, false);
}

// C/D 16x16 f32: VGPR j -> M = j + (lane<16 ? 0 : 8), N = lane&15
__device__ __forceinline__ int acc_row0(int row_base, int lane) {
    return row_base + ((lane >> 4) << 3);
}

// ---------------------------------------------------------------------------
// 1) Bilinear resize (align_corners) 56x56 -> 28x28, f32 -> bf16
// ---------------------------------------------------------------------------
__device__ __forceinline__ float bilerp56(const float* img, int oh, int ow) {
    const float s = 55.0f / 27.0f;
    float ys = oh * s, xs = ow * s;
    int y0 = (int)ys, x0 = (int)xs;
    int y1 = min(y0 + 1, 55), x1 = min(x0 + 1, 55);
    float wy = ys - y0, wx = xs - x0;
    float a = img[y0 * HS + x0], b = img[y0 * HS + x1];
    float c = img[y1 * HS + x0], d = img[y1 * HS + x1];
    return (a * (1.f - wx) + b * wx) * (1.f - wy) + (c * (1.f - wx) + d * wx) * wy;
}

// f1t: (B, P2, 256) row-major (A operand); rows >=784 left uninitialized (pad)
__global__ void resize_f1t_kernel(const float* __restrict__ f1, __bf16* __restrict__ f1t) {
    int idx = blockIdx.x * blockDim.x + threadIdx.x;
    if (idx >= B_ * PQ * C_) return;
    int c = idx % C_, p = (idx / C_) % PQ, b = idx / (C_ * PQ);
    const float* img = f1 + ((size_t)b * C_ + c) * HS * HS;
    f1t[((size_t)b * P2 + p) * C_ + c] = (__bf16)bilerp56(img, p / OH, p % OH);
}

// f2sw: (B, 8 ktiles, 800 cols, 32) swizzled B operand
__global__ void resize_f2sw_kernel(const float* __restrict__ f2, __bf16* __restrict__ f2sw) {
    int idx = blockIdx.x * blockDim.x + threadIdx.x;
    if (idx >= B_ * C_ * PQ) return;
    int q = idx % PQ, c = (idx / PQ) % C_, b = idx / (PQ * C_);
    const float* img = f2 + ((size_t)b * C_ + c) * HS * HS;
    f2sw[(((size_t)b * KTC + (c >> 5)) * P2 + q) * 32 + (c & 31)]
        = (__bf16)bilerp56(img, q / OH, q % OH);
}

// 2) mask maxpool 2x2/2 -> (B,1,28,28)
__global__ void maskpool_kernel(const float* __restrict__ mask, float* __restrict__ m) {
    int idx = blockIdx.x * blockDim.x + threadIdx.x;
    if (idx >= B_ * PQ) return;
    int q = idx % PQ, b = idx / PQ;
    int h = q / OH, w = q % OH;
    const float* src = mask + (size_t)b * HS * HS + (2 * h) * HS + 2 * w;
    m[idx] = fmaxf(fmaxf(src[0], src[1]), fmaxf(src[HS], src[HS + 1]));
}

// ---------------------------------------------------------------------------
// 3) Correlation GEMM: corr[p,q] = sum_c f1t[p,c]*f2[c,q], ReLU
//    32x32 tile per wave (2x2 WMMA), 4 waves/block over N-tiles.
//    Output written directly in swizzled B-operand layout (25,800,32).
// ---------------------------------------------------------------------------
__global__ __launch_bounds__(128)
void corr_gemm_kernel(const __bf16* __restrict__ f1t, const __bf16* __restrict__ f2sw,
                      __bf16* __restrict__ corrn) {
    int lane = threadIdx.x & 31;
    int nt = (blockIdx.x * 4 + (threadIdx.x >> 5)) * 32;   // N tile base
    if (nt >= P2) return;
    int mt = blockIdx.y * 32, b = blockIdx.z;
    const __bf16* A  = f1t + (size_t)b * P2 * C_;
    const __bf16* Bs = f2sw + (size_t)b * KTC * P2 * 32;
    v8f acc00 = {}, acc01 = {}, acc10 = {}, acc11 = {};
#pragma unroll
    for (int k0 = 0; k0 < C_; k0 += 32) {
        __builtin_prefetch(A + (size_t)(mt + (lane & 15)) * C_ + k0 + 32, 0, 0);
        v16bf a0 = load_a_frag(A, C_, mt,      k0, lane);
        v16bf a1 = load_a_frag(A, C_, mt + 16, k0, lane);
        v16bf b0 = load_b_sw(Bs, P2, k0, nt,      lane);
        v16bf b1 = load_b_sw(Bs, P2, k0, nt + 16, lane);
        acc00 = wmma_bf16(a0, b0, acc00);
        acc01 = wmma_bf16(a0, b1, acc01);
        acc10 = wmma_bf16(a1, b0, acc10);
        acc11 = wmma_bf16(a1, b1, acc11);
    }
    __bf16* Cp = corrn + (size_t)b * KT1 * P2 * 32;
    v8f accs[4] = {acc00, acc01, acc10, acc11};
#pragma unroll
    for (int t = 0; t < 4; ++t) {
        int m0 = acc_row0(mt + (t >> 1) * 16, lane);
        int n  = nt + (t & 1) * 16 + (lane & 15);
        v8bf st;
#pragma unroll
        for (int j = 0; j < 8; ++j) {
            float v = accs[t][j];
            st[j] = (__bf16)(v > 0.f ? v : 0.f);
        }
        // swizzled store: rows m0..m0+7 are contiguous kk of tile m0>>5
        *(v8bf*)(Cp + ((size_t)(m0 >> 5) * P2 + n) * 32 + (m0 & 31)) = st;
    }
}

// 4) L2 norm over p (swizzled layout) -> scale[b,q] = mask / sqrt(sum corr^2)
__global__ void norm_kernel(const __bf16* __restrict__ corrn, const float* __restrict__ m,
                            float* __restrict__ scale) {
    int q = blockIdx.x * blockDim.x + threadIdx.x;
    int b = blockIdx.y;
    if (q >= PQ) return;
    const __bf16* Cp = corrn + (size_t)b * KT1 * P2 * 32 + (size_t)q * 32;
    float s = 0.f;
    for (int kt = 0; kt < KT1; ++kt) {
        int kmax = min(32, PQ - kt * 32);           // last tile: only 16 real rows
        const __bf16* chunk = Cp + (size_t)kt * P2 * 32;
        for (int kk = 0; kk < kmax; ++kk) {
            float v = (float)chunk[kk];
            s += v * v;
        }
    }
    scale[b * PQ + q] = m[b * PQ + q] * rsqrtf(s);
}

// 5) corr *= scale (in place); pad rows (p>=784) forced to zero
__global__ void normalize_kernel(__bf16* __restrict__ corrn, const float* __restrict__ scale) {
    size_t idx = (size_t)blockIdx.x * blockDim.x + threadIdx.x;
    if (idx >= (size_t)B_ * KT1 * PQ * 32) return;
    int kk = idx & 31;
    int q  = (idx >> 5) % PQ;
    int kt = (idx / ((size_t)32 * PQ)) % KT1;
    int b  = idx / ((size_t)32 * PQ * KT1);
    size_t a = (((size_t)b * KT1 + kt) * P2 + q) * 32 + kk;
    int p = kt * 32 + kk;
    if (p < PQ) {
        corrn[a] = (__bf16)((float)corrn[a] * scale[b * PQ + q]);
    } else {
        corrn[a] = (__bf16)0.f;
    }
}

// 6) 3x3/1/pad1 spatial maxpool per p-channel (swizzled in/out)
__global__ void pool3x3_kernel(const __bf16* __restrict__ corrn, __bf16* __restrict__ pooled) {
    size_t idx = (size_t)blockIdx.x * blockDim.x + threadIdx.x;
    if (idx >= (size_t)B_ * KT1 * PQ * 32) return;
    int kk = idx & 31;
    int q  = (idx >> 5) % PQ;
    int kt = (idx / ((size_t)32 * PQ)) % KT1;
    int b  = idx / ((size_t)32 * PQ * KT1);
    size_t rowBase = ((size_t)b * KT1 + kt) * P2 * 32;
    size_t a = rowBase + (size_t)q * 32 + kk;
    int p = kt * 32 + kk;
    if (p >= PQ) { pooled[a] = (__bf16)0.f; return; }
    int h = q / OH, w = q % OH;
    float mx = -1e30f;
#pragma unroll
    for (int dy = -1; dy <= 1; ++dy) {
        int hh = h + dy; if (hh < 0 || hh >= OH) continue;
#pragma unroll
        for (int dx = -1; dx <= 1; ++dx) {
            int ww = w + dx; if (ww < 0 || ww >= OH) continue;
            mx = fmaxf(mx, (float)corrn[rowBase + (size_t)(hh * OH + ww) * 32 + kk]);
        }
    }
    pooled[a] = (__bf16)mx;
}

// ---------------------------------------------------------------------------
// Weight packing
// ---------------------------------------------------------------------------
__global__ void pack_w1x1_kernel(const float* __restrict__ Ws, __bf16* __restrict__ Wd, int OC) {
    int idx = blockIdx.x * blockDim.x + threadIdx.x;
    if (idx >= OC * P2) return;
    int oc = idx / P2, k = idx % P2;
    Wd[idx] = (__bf16)(k < PQ ? Ws[oc * PQ + k] : 0.f);
}
__global__ void pack_wKxK_kernel(const float* __restrict__ Ws, __bf16* __restrict__ Wd,
                                 int OC, int CIN, int KK) { // Wd[t][oc][c] = Ws[oc][c][t]
    int idx = blockIdx.x * blockDim.x + threadIdx.x;
    if (idx >= KK * OC * CIN) return;
    int t = idx / (OC * CIN), r = idx % (OC * CIN);
    int oc = r / CIN, c = r % CIN;
    Wd[idx] = (__bf16)Ws[((size_t)oc * CIN + c) * KK + t];
}

// ---------------------------------------------------------------------------
// 7) 1x1 conv as GEMM: Y = relu(W(OC,800) x In_swizzled(800,800) + bias)
//    mode 0: f32 -> y (d_out) at channel offset (guard n<784)
//    mode 1: bf16 -> channel-swizzled zero-padded intermediate
//            layout (B, CIN/32, padH*padW, 32), pixel offset +pr border
// ---------------------------------------------------------------------------
__global__ __launch_bounds__(128)
void conv1x1_gemm_kernel(const __bf16* __restrict__ Wp, const __bf16* __restrict__ InSw,
                         const float* __restrict__ bias,
                         float* __restrict__ outY, __bf16* __restrict__ outSw,
                         int mode, int chOff, int padW, int pr) {
    int lane = threadIdx.x & 31;
    int nt = (blockIdx.x * 4 + (threadIdx.x >> 5)) * 32;
    if (nt >= P2) return;
    int mt = blockIdx.y * 32, b = blockIdx.z;
    const __bf16* Bs = InSw + (size_t)b * KT1 * P2 * 32;
    v8f acc00 = {}, acc01 = {}, acc10 = {}, acc11 = {};
    for (int k0 = 0; k0 < P2; k0 += 32) {
        __builtin_prefetch(Wp + (size_t)(mt + (lane & 15)) * P2 + k0 + 32, 0, 0);
        v16bf a0 = load_a_frag(Wp, P2, mt,      k0, lane);
        v16bf a1 = load_a_frag(Wp, P2, mt + 16, k0, lane);
        v16bf b0 = load_b_sw(Bs, P2, k0, nt,      lane);
        v16bf b1 = load_b_sw(Bs, P2, k0, nt + 16, lane);
        acc00 = wmma_bf16(a0, b0, acc00);
        acc01 = wmma_bf16(a0, b1, acc01);
        acc10 = wmma_bf16(a1, b0, acc10);
        acc11 = wmma_bf16(a1, b1, acc11);
    }
    v8f accs[4] = {acc00, acc01, acc10, acc11};
#pragma unroll
    for (int t = 0; t < 4; ++t) {
        int m0 = acc_row0(mt + (t >> 1) * 16, lane);
        int n  = nt + (t & 1) * 16 + (lane & 15);
        if (n >= PQ) continue;                       // pad columns discarded
        if (mode == 0) {
            float* Yp = outY + (size_t)b * OC_ALL * PQ + (size_t)chOff * PQ;
#pragma unroll
            for (int j = 0; j < 8; ++j) {
                float v = accs[t][j] + bias[m0 + j];
                Yp[(size_t)(m0 + j) * PQ + n] = v > 0.f ? v : 0.f;
            }
        } else {
            int h = n / OH, w = n % OH;
            int pix = (h + pr) * padW + (w + pr);
            int CINt = gridDim.y * 2;                // 2*(OC/32) = OC/16 channel tiles per batch stride
            __bf16* Pp = outSw + (size_t)b * CINt * padW * padW * 32;
            v8bf st;
#pragma unroll
            for (int j = 0; j < 8; ++j) {
                float v = accs[t][j] + bias[m0 + j];
                st[j] = (__bf16)(v > 0.f ? v : 0.f);
            }
            *(v8bf*)(Pp + ((size_t)(m0 >> 5) * padW * padW + pix) * 32 + (m0 & 31)) = st;
        }
    }
}

// ---------------------------------------------------------------------------
// 8) KxK conv: shift-accumulate WMMA over channel-swizzled padded input
//    Wp (KK, OC, CIN) row-major; input (B, CIN/32, PW*PW, 32)
// ---------------------------------------------------------------------------
template <int KS, int CIN, int PW>
__global__ __launch_bounds__(128)
void convKxK_kernel(const __bf16* __restrict__ Wp, const __bf16* __restrict__ inSw,
                    const float* __restrict__ bias, float* __restrict__ outY, int chOff) {
    int lane = threadIdx.x & 31;
    int nt = (blockIdx.x * 4 + (threadIdx.x >> 5)) * 32;
    if (nt >= P2) return;
    int mt = blockIdx.y * 32, b = blockIdx.z;
    const int PIX = PW * PW, NKT = CIN / 32;
    // per-lane pixel coords for both column halves (pad columns clamped; stores guarded)
    int n0 = nt + (lane & 15), n1 = n0 + 16;
    int q0 = min(n0, PQ - 1), q1 = min(n1, PQ - 1);
    int h0 = q0 / OH, w0 = q0 % OH, h1 = q1 / OH, w1 = q1 % OH;
    const __bf16* inB = inSw + (size_t)b * NKT * PIX * 32;
    v8f acc00 = {}, acc01 = {}, acc10 = {}, acc11 = {};
#pragma unroll
    for (int t = 0; t < KS * KS; ++t) {
        int dy = t / KS, dx = t % KS;
        const __bf16* At = Wp + (size_t)t * (gridDim.y * 32) * CIN;
        int pix0 = (h0 + dy) * PW + (w0 + dx);
        int pix1 = (h1 + dy) * PW + (w1 + dx);
        for (int k0 = 0; k0 < CIN; k0 += 32) {
            v16bf a0 = load_a_frag(At, CIN, mt,      k0, lane);
            v16bf a1 = load_a_frag(At, CIN, mt + 16, k0, lane);
            int kh = (lane >> 4) << 4;
            const __bf16* pB0 = inB + ((size_t)(k0 >> 5) * PIX + pix0) * 32 + kh;
            const __bf16* pB1 = inB + ((size_t)(k0 >> 5) * PIX + pix1) * 32 + kh;
            v8bf l0 = *(const v8bf*)pB0, hpart0 = *(const v8bf*)(pB0 + 8);
            v8bf l1 = *(const v8bf*)pB1, hpart1 = *(const v8bf*)(pB1 + 8);
            v16bf b0, b1;
#pragma unroll
            for (int i = 0; i < 8; ++i) {
                b0[i] = l0[i]; b0[i + 8] = hpart0[i];
                b1[i] = l1[i]; b1[i + 8] = hpart1[i];
            }
            acc00 = wmma_bf16(a0, b0, acc00);
            acc01 = wmma_bf16(a0, b1, acc01);
            acc10 = wmma_bf16(a1, b0, acc10);
            acc11 = wmma_bf16(a1, b1, acc11);
        }
    }
    float* Yp = outY + (size_t)b * OC_ALL * PQ + (size_t)chOff * PQ;
    v8f accs[4] = {acc00, acc01, acc10, acc11};
#pragma unroll
    for (int t = 0; t < 4; ++t) {
        int m0 = acc_row0(mt + (t >> 1) * 16, lane);
        int n  = nt + (t & 1) * 16 + (lane & 15);
        if (n >= PQ) continue;
#pragma unroll
        for (int j = 0; j < 8; ++j) {
            float v = accs[t][j] + bias[m0 + j];
            Yp[(size_t)(m0 + j) * PQ + n] = v > 0.f ? v : 0.f;
        }
    }
}

// ---------------------------------------------------------------------------
// 9) BatchNorm: per-channel stats over (B,28,28), then in-place apply
// ---------------------------------------------------------------------------
__global__ void bnstats_kernel(const float* __restrict__ y,
                               float* __restrict__ meanA, float* __restrict__ istdA) {
    int ch = blockIdx.x;
    float s = 0.f, ss = 0.f;
    for (int i = threadIdx.x; i < B_ * PQ; i += blockDim.x) {
        int b = i / PQ, q = i % PQ;
        float v = y[((size_t)b * OC_ALL + ch) * PQ + q];
        s += v; ss += v * v;
    }
    __shared__ float rs[256], rss[256];
    rs[threadIdx.x] = s; rss[threadIdx.x] = ss;
    __syncthreads();
    for (int off = 128; off > 0; off >>= 1) {
        if (threadIdx.x < off) {
            rs[threadIdx.x]  += rs[threadIdx.x + off];
            rss[threadIdx.x] += rss[threadIdx.x + off];
        }
        __syncthreads();
    }
    if (threadIdx.x == 0) {
        float mu  = rs[0] / (float)(B_ * PQ);
        float var = rss[0] / (float)(B_ * PQ) - mu * mu;
        meanA[ch] = mu;
        istdA[ch] = rsqrtf(var + 1e-5f);
    }
}

__global__ void bnapply_kernel(float* __restrict__ y,
                               const float* __restrict__ meanA, const float* __restrict__ istdA,
                               const float* __restrict__ gamma, const float* __restrict__ beta) {
    size_t idx = (size_t)blockIdx.x * blockDim.x + threadIdx.x;
    if (idx >= (size_t)B_ * OC_ALL * PQ) return;
    int ch = (idx / PQ) % OC_ALL;
    y[idx] = gamma[ch] * (y[idx] - meanA[ch]) * istdA[ch] + beta[ch];
}

// ---------------------------------------------------------------------------
// Launcher
// ---------------------------------------------------------------------------
static inline size_t alignUp(size_t v) { return (v + 255) & ~(size_t)255; }

extern "C" void kernel_launch(void* const* d_in, const int* in_sizes, int n_in,
                              void* d_out, int out_size, void* d_ws, size_t ws_size,
                              hipStream_t stream) {
    const float* feature1 = (const float*)d_in[0];
    const float* feature2 = (const float*)d_in[1];
    const float* mask     = (const float*)d_in[2];
    const float* W1  = (const float*)d_in[3];  const float* b1  = (const float*)d_in[4];
    const float* W2a = (const float*)d_in[5];  const float* b2a = (const float*)d_in[6];
    const float* W2b = (const float*)d_in[7];  const float* b2b = (const float*)d_in[8];
    const float* W3a = (const float*)d_in[9];  const float* b3a = (const float*)d_in[10];
    const float* W3b = (const float*)d_in[11]; const float* b3b = (const float*)d_in[12];
    const float* W4  = (const float*)d_in[13]; const float* b4  = (const float*)d_in[14];
    const float* gamma = (const float*)d_in[15];
    const float* beta  = (const float*)d_in[16];
    float* y = (float*)d_out;

    // workspace carve
    char* ws = (char*)d_ws; size_t off = 0;
    auto carve = [&](size_t bytes) { void* p = ws + off; off += alignUp(bytes); return p; };
    __bf16* f1t    = (__bf16*)carve((size_t)B_ * P2 * C_ * 2);          // A operand (rows padded)
    __bf16* f2sw   = (__bf16*)carve((size_t)B_ * KTC * P2 * 32 * 2);    // swizzled B
    __bf16* corrn  = (__bf16*)carve((size_t)B_ * KT1 * P2 * 32 * 2);    // swizzled corr
    __bf16* pooled = (__bf16*)carve((size_t)B_ * KT1 * P2 * 32 * 2);    // swizzled pooled corr
    float*  mp     = (float*) carve((size_t)B_ * PQ * 4);
    float*  scale  = (float*) carve((size_t)B_ * PQ * 4);
    __bf16* y2aSw  = (__bf16*)carve((size_t)B_ * 6 * 30 * 30 * 32 * 2); // (B,192/32,900,32)
    __bf16* y3aSw  = (__bf16*)carve((size_t)B_ * 1 * 32 * 32 * 32 * 2); // (B,32/32,1024,32)
    __bf16* Wp1    = (__bf16*)carve((size_t)128 * P2 * 2);
    __bf16* Wp2a   = (__bf16*)carve((size_t)192 * P2 * 2);
    __bf16* Wp3a   = (__bf16*)carve((size_t)32 * P2 * 2);
    __bf16* Wp4    = (__bf16*)carve((size_t)64 * P2 * 2);
    __bf16* Wp2b   = (__bf16*)carve((size_t)9 * 256 * 192 * 2);
    __bf16* Wp3b   = (__bf16*)carve((size_t)25 * 64 * 32 * 2);
    float*  meanA  = (float*) carve(512 * 4);
    float*  istdA  = (float*) carve(512 * 4);
    (void)ws_size; (void)in_sizes; (void)n_in; (void)out_size;

    // zero-init padded conv intermediates (borders must be zero)
    hipMemsetAsync(y2aSw, 0, (size_t)B_ * 6 * 30 * 30 * 32 * 2, stream);
    hipMemsetAsync(y3aSw, 0, (size_t)B_ * 1 * 32 * 32 * 32 * 2, stream);

    const int T = 256;
    // preprocessing
    resize_f1t_kernel<<<(B_ * PQ * C_ + T - 1) / T, T, 0, stream>>>(feature1, f1t);
    resize_f2sw_kernel<<<(B_ * C_ * PQ + T - 1) / T, T, 0, stream>>>(feature2, f2sw);
    maskpool_kernel<<<(B_ * PQ + T - 1) / T, T, 0, stream>>>(mask, mp);

    // weight packs
    pack_w1x1_kernel<<<(128 * P2 + T - 1) / T, T, 0, stream>>>(W1,  Wp1, 128);
    pack_w1x1_kernel<<<(192 * P2 + T - 1) / T, T, 0, stream>>>(W2a, Wp2a, 192);
    pack_w1x1_kernel<<<(32  * P2 + T - 1) / T, T, 0, stream>>>(W3a, Wp3a, 32);
    pack_w1x1_kernel<<<(64  * P2 + T - 1) / T, T, 0, stream>>>(W4,  Wp4, 64);
    pack_wKxK_kernel<<<(9 * 256 * 192 + T - 1) / T, T, 0, stream>>>(W2b, Wp2b, 256, 192, 9);
    pack_wKxK_kernel<<<(25 * 64 * 32 + T - 1) / T, T, 0, stream>>>(W3b, Wp3b, 64, 32, 25);

    // correlation GEMM + relu (25 N-tiles -> 7 blocks x 4 waves, 25 M-tiles)
    corr_gemm_kernel<<<dim3(7, 25, B_), 128, 0, stream>>>(f1t, f2sw, corrn);

    // L2 norm scale, in-place normalize (+ zero K-pad rows), 3x3 pool
    norm_kernel<<<dim3((PQ + T - 1) / T, B_), T, 0, stream>>>(corrn, mp, scale);
    size_t nElem = (size_t)B_ * KT1 * PQ * 32;
    normalize_kernel<<<(nElem + T - 1) / T, T, 0, stream>>>(corrn, scale);
    pool3x3_kernel<<<(nElem + T - 1) / T, T, 0, stream>>>(corrn, pooled);

    // inception branches
    conv1x1_gemm_kernel<<<dim3(7, 4, B_), 128, 0, stream>>>(   // y1 -> y[:,0:128]
        Wp1, corrn, b1, y, nullptr, 0, 0, 0, 0);
    conv1x1_gemm_kernel<<<dim3(7, 6, B_), 128, 0, stream>>>(   // y2a -> swizzled (B,6,900,32)
        Wp2a, corrn, b2a, nullptr, y2aSw, 1, 0, 30, 1);
    conv1x1_gemm_kernel<<<dim3(7, 1, B_), 128, 0, stream>>>(   // y3a -> swizzled (B,1,1024,32)
        Wp3a, corrn, b3a, nullptr, y3aSw, 1, 0, 32, 2);
    conv1x1_gemm_kernel<<<dim3(7, 2, B_), 128, 0, stream>>>(   // y4 -> y[:,448:512]
        Wp4, pooled, b4, y, nullptr, 0, 448, 0, 0);
    convKxK_kernel<3, 192, 30><<<dim3(7, 8, B_), 128, 0, stream>>>(Wp2b, y2aSw, b2b, y, 128);
    convKxK_kernel<5, 32, 32><<<dim3(7, 2, B_), 128, 0, stream>>>(Wp3b, y3aSw, b3b, y, 384);

    // batchnorm (training stats), in-place on d_out
    bnstats_kernel<<<512, 256, 0, stream>>>(y, meanA, istdA);
    size_t yElem = (size_t)B_ * OC_ALL * PQ;
    bnapply_kernel<<<(yElem + T - 1) / T, T, 0, stream>>>(y, meanA, istdA, gamma, beta);
}
